// MultiHeadAttention_74191265071493
// MI455X (gfx1250) — compile-verified, matrix operand order checked by reading, not run on hardware
//
#include <hip/hip_runtime.h>

typedef __attribute__((ext_vector_type(16))) _Float16     v16h;
typedef __attribute__((ext_vector_type(8)))  _Float16     v8h;
typedef __attribute__((ext_vector_type(8)))  float        v8f;
typedef __attribute__((ext_vector_type(4)))  float        v4f;
typedef __attribute__((ext_vector_type(4)))  unsigned int v4u;
typedef __attribute__((ext_vector_type(8)))  int          v8i;
typedef __attribute__((ext_vector_type(4)))  int          v4i;

#define D_MODEL 1024
#define N_HEADS 16
#define D_HEAD  64
#define SEQ     2048
#define BATCH   4
#define M_TOK   (BATCH * SEQ)   // 8192

#if defined(__has_builtin)
#  if __has_builtin(__builtin_amdgcn_tensor_load_to_lds) && \
      __has_builtin(__builtin_amdgcn_s_wait_tensorcnt)
#    define USE_TDM 1
#  endif
#endif
#ifndef USE_TDM
#  define USE_TDM 0
#endif

// ---------------------------------------------------------------------------
// Fragment loaders: 16x32 f16 tile in WMMA A/B layout (wave32).
// Lane l: row = l&15 ; K-chunks at k0 = (l>>4)*8 and 16 + k0 (8 halfs each).
// Works for A (lane = M row) and for B when given B^T row-major (lane = N col).
// ---------------------------------------------------------------------------
__device__ __forceinline__ v16h load_frag_h(const _Float16* base, int ld) {
    const int lane = threadIdx.x & 31;
    const int row  = lane & 15;
    const int k0   = (lane >> 4) << 3;          // 0 or 8
    const _Float16* p = base + (size_t)row * ld;
    v8h lo = *(const v8h*)(p + k0);
    v8h hi = *(const v8h*)(p + 16 + k0);
    v16h r;
#pragma unroll
    for (int i = 0; i < 8; ++i) { r[i] = lo[i]; r[8 + i] = hi[i]; }
    return r;
}

__device__ __forceinline__ v16h load_frag_f(const float* __restrict__ base, int ld) {
    const int lane = threadIdx.x & 31;
    const int row  = lane & 15;
    const int k0   = (lane >> 4) << 3;
    const float* p = base + (size_t)row * ld;
    v4f a0 = *(const v4f*)(p + k0);
    v4f a1 = *(const v4f*)(p + k0 + 4);
    v4f b0 = *(const v4f*)(p + 16 + k0);
    v4f b1 = *(const v4f*)(p + 16 + k0 + 4);
    v16h r;
#pragma unroll
    for (int i = 0; i < 4; ++i) {
        r[i]      = (_Float16)a0[i];
        r[4 + i]  = (_Float16)a1[i];
        r[8 + i]  = (_Float16)b0[i];
        r[12 + i] = (_Float16)b1[i];
    }
    return r;
}

__device__ __forceinline__ v16h load_any(const float* p, int ld)    { return load_frag_f(p, ld); }
__device__ __forceinline__ v16h load_any(const _Float16* p, int ld) { return load_frag_h(p, ld); }

#define WMMA_F16(A, B, C) \
    __builtin_amdgcn_wmma_f32_16x16x32_f16(false, (A), false, (B), (short)0, (C), false, false)

#if USE_TDM
// ---------------------------------------------------------------------------
// TDM: 2-D tile load Global->LDS via Tensor Data Mover (ISA 08, D# §8.3-8.5).
// data_size = 2 bytes (f16). tile_d0 = contiguous elements per row,
// tile_d1 = rows, stride_d0 = row stride (elements). One instruction/wave,
// tracked by TENSORcnt. 6-arg builtin form (clang-23 toolchain).
// ---------------------------------------------------------------------------
__device__ __forceinline__ void tdm_load_2d(const _Float16* gaddr, unsigned lds_off,
                                            unsigned tile_d0, unsigned tile_d1,
                                            unsigned tensor_d0, unsigned tensor_d1,
                                            unsigned long long stride_d0) {
    const unsigned long long ga = (unsigned long long)(size_t)gaddr;
    v4u g0;
    g0[0] = 1u;                                         // count=1, user mode
    g0[1] = lds_off;                                    // lds_addr (bytes)
    g0[2] = (unsigned)(ga & 0xffffffffu);               // global_addr[31:0]
    g0[3] = (unsigned)((ga >> 32) & 0x01ffffffu)        // global_addr[56:32]
            | (2u << 30);                               // type = 2 (image)
    v8i g1;
    g1[0] = (int)(1u << 16);                            // data_size = 1 -> 2B
    g1[1] = (int)((tensor_d0 & 0xffffu) << 16);         // tensor_dim0[15:0]
    g1[2] = (int)(((tensor_d0 >> 16) & 0xffffu) |
                  ((tensor_d1 & 0xffffu) << 16));       // dim0[31:16] | dim1[15:0]
    g1[3] = (int)(((tensor_d1 >> 16) & 0xffffu) |
                  ((tile_d0 & 0xffffu) << 16));         // dim1[31:16] | tile_dim0
    g1[4] = (int)(tile_d1 & 0xffffu);                   // tile_dim1 ; tile_dim2=0
    g1[5] = (int)(stride_d0 & 0xffffffffull);           // dim0_stride[31:0]
    g1[6] = (int)((stride_d0 >> 32) & 0xffffull);       // dim0_stride[47:32]
    g1[7] = 0;                                          // dim1_stride unused (2D)
    v4i z4 = (v4i){};
    v8i z8 = (v8i){};
    __builtin_amdgcn_tensor_load_to_lds(g0, g1, z4, z4, z8, 0);
}
#endif

// ---------------------------------------------------------------------------
// fp32 -> f16 elementwise (weight conversion), n multiple of 4.
// ---------------------------------------------------------------------------
__global__ __launch_bounds__(256) void f32_to_f16(const float* __restrict__ in,
                                                  _Float16* __restrict__ out, int n) {
    const int i = (blockIdx.x * 256 + threadIdx.x) * 4;
    if (i < n) {
        v4f v = *(const v4f*)(in + i);
#pragma unroll
        for (int j = 0; j < 4; ++j) out[i + j] = (_Float16)v[j];
    }
}

// ---------------------------------------------------------------------------
// GEMM: out[m,n] = sum_k A[m,k] * W[n,k]  (A @ W^T), M=8192, N=K=1024, W in f16.
// One wave -> 32x64 output block (2x4 WMMA tiles). 8 waves / block.
// mode 0: f16 store as [B,H,L,Dh]     (Q / K projections, optional scale)
// mode 1: f16 store as [B,H,Dh,L]     (V projection, transposed for P@V)
// mode 2: f32 store row-major [M, N]  (final output projection)
// ---------------------------------------------------------------------------
template <typename AT>
__global__ __launch_bounds__(256) void gemm_wmma(
        const AT* __restrict__ A, const _Float16* __restrict__ W,
        _Float16* __restrict__ outH, float* __restrict__ outF,
        int mode, float scale) {
    const int wid = blockIdx.x * 8 + (threadIdx.x >> 5);
    const int m0  = (wid >> 4) * 32;
    const int n0  = (wid & 15) * 64;

    v8f acc[2][4];
#pragma unroll
    for (int i = 0; i < 2; ++i)
#pragma unroll
        for (int j = 0; j < 4; ++j) acc[i][j] = (v8f){};

    for (int k0 = 0; k0 < D_MODEL; k0 += 32) {
        __builtin_prefetch(A + (size_t)m0 * D_MODEL + k0 + 32, 0, 1);
        v16h a0 = load_any(A + (size_t)m0 * D_MODEL + k0, D_MODEL);
        v16h a1 = load_any(A + (size_t)(m0 + 16) * D_MODEL + k0, D_MODEL);
        v16h b[4];
#pragma unroll
        for (int j = 0; j < 4; ++j)
            b[j] = load_frag_h(W + (size_t)(n0 + 16 * j) * D_MODEL + k0, D_MODEL);
#pragma unroll
        for (int j = 0; j < 4; ++j) {
            acc[0][j] = WMMA_F16(a0, b[j], acc[0][j]);
            acc[1][j] = WMMA_F16(a1, b[j], acc[1][j]);
        }
    }

    const int lane = threadIdx.x & 31;
    const int n    = lane & 15;
    const int mb   = (lane >> 4) * 8;
#pragma unroll
    for (int i = 0; i < 2; ++i)
#pragma unroll
        for (int j = 0; j < 4; ++j)
#pragma unroll
            for (int r = 0; r < 8; ++r) {
                const int m  = m0 + 16 * i + mb + r;
                const int ng = n0 + 16 * j + n;
                const float v = acc[i][j][r] * scale;
                if (mode == 2) {
                    outF[(size_t)m * D_MODEL + ng] = v;
                } else {
                    const int bb = m >> 11, l = m & (SEQ - 1);
                    const int h = ng >> 6, dh = ng & (D_HEAD - 1);
                    if (mode == 0)
                        outH[(((size_t)(bb * N_HEADS + h)) * SEQ + l) * D_HEAD + dh] = (_Float16)v;
                    else
                        outH[(((size_t)(bb * N_HEADS + h)) * D_HEAD + dh) * SEQ + l] = (_Float16)v;
                }
            }
}

// ---------------------------------------------------------------------------
// Causal flash attention. Grid: (SEQ/16, B*H), block: 1 wave (32 threads).
// Q pre-scaled by 1/sqrt(Dh). Key blocks of 32. K/V tiles streamed into LDS
// by the Tensor Data Mover, double-buffered on TENSORcnt; online softmax; P
// staged through LDS to convert C-layout -> A-fragment layout for P@V WMMAs.
// ---------------------------------------------------------------------------
__global__ __launch_bounds__(32) void flash_attn(
        const _Float16* __restrict__ Q, const _Float16* __restrict__ K,
        const _Float16* __restrict__ Vt, const int* __restrict__ amask,
        _Float16* __restrict__ Ctx) {
#if USE_TDM
    __shared__ _Float16 lk[2][32 * 64];   // K tile: 32 keys x 64 dh
    __shared__ _Float16 lv[2][64 * 32];   // V^T tile: 64 dh x 32 keys
#endif
    __shared__ _Float16 lp[16 * 32];      // P tile staging

    const int qt = blockIdx.x;
    const int bh = blockIdx.y;
    const int b  = bh >> 4;          // / N_HEADS
    const int h  = bh & 15;
    const int qb = qt * 16;

    const _Float16* Qb  = Q  + (size_t)bh * SEQ * D_HEAD;
    const _Float16* Kb  = K  + (size_t)bh * SEQ * D_HEAD;
    const _Float16* Vtb = Vt + (size_t)bh * D_HEAD * SEQ;
    const int*      mkb = amask + (size_t)b * SEQ;

    const int lane = threadIdx.x & 31;
    const int n    = lane & 15;
    const int mb   = (lane >> 4) * 8;

    // Q fragments resident for the whole row-tile (Dh = 64 -> 2 K-steps).
    const v16h qf0 = load_frag_h(Qb + (size_t)qb * D_HEAD, D_HEAD);
    const v16h qf1 = load_frag_h(Qb + (size_t)qb * D_HEAD + 32, D_HEAD);

    v8f acc[4];
#pragma unroll
    for (int d = 0; d < 4; ++d) acc[d] = (v8f){};
    float m_r[8], l_r[8];
#pragma unroll
    for (int r = 0; r < 8; ++r) { m_r[r] = -1e30f; l_r[r] = 0.0f; }

    const int nkb = (qb + 16 + 31) >> 5;   // causal: keys [0, qb+16)

#if USE_TDM
    // Prime pipeline: stream first K / V^T tiles into LDS buffer 0.
    tdm_load_2d(Kb, (unsigned)(size_t)&lk[0][0], 64, 32, 64, SEQ, 64);
    tdm_load_2d(Vtb, (unsigned)(size_t)&lv[0][0], 32, 64, SEQ, 64, SEQ);
#endif

    for (int kb = 0; kb < nkb; ++kb) {
        const int jb = kb * 32;
        const _Float16 *kbase, *vbase;
        int kld, vld;
#if USE_TDM
        const int buf = kb & 1;
        if (kb + 1 < nkb) {   // stream next block into the other buffer
            tdm_load_2d(Kb + (size_t)(jb + 32) * D_HEAD,
                        (unsigned)(size_t)&lk[buf ^ 1][0], 64, 32, 64, SEQ, 64);
            tdm_load_2d(Vtb + (jb + 32),
                        (unsigned)(size_t)&lv[buf ^ 1][0], 32, 64, SEQ, 64, SEQ);
            __builtin_amdgcn_s_wait_tensorcnt(2);   // oldest 2 (current buf) done
        } else {
            __builtin_amdgcn_s_wait_tensorcnt(0);
        }
        kbase = &lk[buf][0]; kld = 64;
        vbase = &lv[buf][0]; vld = 32;
#else
        kbase = Kb + (size_t)jb * D_HEAD; kld = D_HEAD;
        vbase = Vtb + jb;                 vld = SEQ;
#endif

        // S = (Q*scale) @ K^T -> two 16x16 tiles (keys jb..jb+15, jb+16..jb+31)
        v16h kf00 = load_frag_h(kbase, kld);
        v16h kf01 = load_frag_h(kbase + 32, kld);
        v16h kf10 = load_frag_h(kbase + 16 * kld, kld);
        v16h kf11 = load_frag_h(kbase + 16 * kld + 32, kld);
        v8f s0 = WMMA_F16(qf0, kf00, (v8f){});
        s0     = WMMA_F16(qf1, kf01, s0);
        v8f s1 = WMMA_F16(qf0, kf10, (v8f){});
        s1     = WMMA_F16(qf1, kf11, s1);

        // Causal + key-validity masking in C layout (lane = column).
        const int kg0 = jb + n, kg1 = jb + 16 + n;
        const bool kv0 = (mkb[kg0] != 0), kv1 = (mkb[kg1] != 0);
        float p0[8], p1[8];
#pragma unroll
        for (int r = 0; r < 8; ++r) {
            const int mg = qb + mb + r;
            p0[r] = ((kg0 > mg) | !kv0) ? -1e30f : s0[r];
            p1[r] = ((kg1 > mg) | !kv1) ? -1e30f : s1[r];
        }

        // Online softmax: row stats via 16-lane xor reductions.
#pragma unroll
        for (int r = 0; r < 8; ++r) {
            float cur = fmaxf(p0[r], p1[r]);
#pragma unroll
            for (int off = 1; off < 16; off <<= 1)
                cur = fmaxf(cur, __shfl_xor(cur, off, 32));
            const float mnew = fmaxf(m_r[r], cur);
            const float sc   = __expf(m_r[r] - mnew);
            p0[r] = __expf(p0[r] - mnew);
            p1[r] = __expf(p1[r] - mnew);
            float rs = p0[r] + p1[r];
#pragma unroll
            for (int off = 1; off < 16; off <<= 1)
                rs += __shfl_xor(rs, off, 32);
            l_r[r] = l_r[r] * sc + rs;
            m_r[r] = mnew;
#pragma unroll
            for (int d = 0; d < 4; ++d) acc[d][r] *= sc;
        }

        // P (C layout) -> LDS -> A-fragment layout.
#pragma unroll
        for (int r = 0; r < 8; ++r) {
            lp[(mb + r) * 32 + n]      = (_Float16)p0[r];
            lp[(mb + r) * 32 + 16 + n] = (_Float16)p1[r];
        }
        asm volatile("s_wait_dscnt 0" ::: "memory");
        const v16h pf = load_frag_h(lp, 32);

        // ctx += P @ V : B fragments from V^T rows (contiguous along seq).
#pragma unroll
        for (int d = 0; d < 4; ++d) {
            v16h vf = load_frag_h(vbase + (size_t)(d * 16) * vld, vld);
            acc[d] = WMMA_F16(pf, vf, acc[d]);
        }
        asm volatile("s_wait_dscnt 0" ::: "memory");   // lp reusable next iter
    }

    // Normalize and store ctx as f16 [M_TOK, D_MODEL] for the output GEMM.
#pragma unroll
    for (int r = 0; r < 8; ++r) {
        const float inv = 1.0f / l_r[r];
        const size_t row = ((size_t)b * SEQ + qb + mb + r) * D_MODEL + h * D_HEAD;
#pragma unroll
        for (int d = 0; d < 4; ++d)
            Ctx[row + d * 16 + n] = (_Float16)(acc[d][r] * inv);
    }
}

// ---------------------------------------------------------------------------
extern "C" void kernel_launch(void* const* d_in, const int* in_sizes, int n_in,
                              void* d_out, int out_size, void* d_ws, size_t ws_size,
                              hipStream_t stream) {
    (void)in_sizes; (void)n_in; (void)out_size; (void)ws_size;
    const float* q    = (const float*)d_in[0];
    const float* k    = (const float*)d_in[1];
    const float* v    = (const float*)d_in[2];
    const int*   mask = (const int*)d_in[3];
    const float* Wq   = (const float*)d_in[4];
    const float* Wk   = (const float*)d_in[5];
    const float* Wv   = (const float*)d_in[6];
    const float* Wo   = (const float*)d_in[7];
    float* out = (float*)d_out;

    const size_t SZ = (size_t)M_TOK * D_MODEL;    // 8M elems per f16 tensor
    const size_t WZ = (size_t)D_MODEL * D_MODEL;  // 1M elems per weight
    _Float16* ws  = (_Float16*)d_ws;
    _Float16* Qh  = ws;             // [B,H,L,Dh]
    _Float16* Kh  = ws + SZ;        // [B,H,L,Dh]
    _Float16* Vt  = ws + 2 * SZ;    // [B,H,Dh,L]
    _Float16* Ctx = ws + 3 * SZ;    // [M_TOK, D_MODEL]
    _Float16* Wqh = ws + 4 * SZ;
    _Float16* Wkh = Wqh + WZ;
    _Float16* Wvh = Wkh + WZ;
    _Float16* Woh = Wvh + WZ;

    // Convert weights to f16 once (4 MB each, trivially bandwidth-bound).
    const int cvtBlocks = (int)(WZ / (256 * 4));
    f32_to_f16<<<cvtBlocks, 256, 0, stream>>>(Wq, Wqh, (int)WZ);
    f32_to_f16<<<cvtBlocks, 256, 0, stream>>>(Wk, Wkh, (int)WZ);
    f32_to_f16<<<cvtBlocks, 256, 0, stream>>>(Wv, Wvh, (int)WZ);
    f32_to_f16<<<cvtBlocks, 256, 0, stream>>>(Wo, Woh, (int)WZ);

    const dim3 gGemm(512), bGemm(256);
    const float kscale = 0.125f;   // 1/sqrt(D_HEAD), folded into Q

    gemm_wmma<float><<<gGemm, bGemm, 0, stream>>>(q, Wqh, Qh, nullptr, 0, kscale);
    gemm_wmma<float><<<gGemm, bGemm, 0, stream>>>(k, Wkh, Kh, nullptr, 0, 1.0f);
    gemm_wmma<float><<<gGemm, bGemm, 0, stream>>>(v, Wvh, Vt, nullptr, 1, 1.0f);

    flash_attn<<<dim3(SEQ / 16, BATCH * N_HEADS), 32, 0, stream>>>(Qh, Kh, Vt, mask, Ctx);

    gemm_wmma<_Float16><<<gGemm, bGemm, 0, stream>>>(Ctx, Woh, nullptr, out, 2, 1.0f);
}
